// _CrissCrossAttention3D_4020089389690
// MI455X (gfx1250) — compile-verified, matrix operand order checked by reading, not run on hardware
//
#include <hip/hip_runtime.h>
#include <hip/hip_bf16.h>
#include <stdint.h>

// ---------------- problem constants ----------------
#define NB   2          // batch
#define CC   256        // channels
#define CI   64         // inner channels (q,k)
#define TT   16
#define HH   96
#define WW   96
#define HWD  (HH*WW)            // 9216
#define THW  (TT*HH*WW)         // 147456
#define SS   (TT+HH+WW)         // 208
#define NEGV (-1e30f)

typedef __attribute__((ext_vector_type(16))) __bf16 v16bf;
typedef __attribute__((ext_vector_type(8)))  __bf16 v8bf;
typedef __attribute__((ext_vector_type(8)))  float  v8f;

// K index held by this lane for fragment element i (A 16x32 / B 32x16, bf16)
__device__ __forceinline__ int kidx(int i, int half) {
  return (i < 8) ? (half * 8 + i) : (8 + half * 8 + i);   // i>=8 -> 16 + half*8 + (i-8)
}

__device__ __forceinline__ v8f wmma_bf16(v16bf a, v16bf b, v8f c) {
  return __builtin_amdgcn_wmma_f32_16x16x32_bf16(false, a, false, b, (short)0, c, false, false);
}

// Load a 16-element lane fragment from a contiguous bf16 row (two aligned 16B chunks).
// Works for both global and LDS rows (compiler emits *_load_b128 / ds_load_b128).
__device__ __forceinline__ v16bf load_row_frag(const __bf16* row, int half) {
  v16bf f;
  const __bf16* p0 = row + half * 8;
  const __bf16* p1 = row + 16 + half * 8;
#pragma unroll
  for (int i = 0; i < 8; ++i) { f[i] = p0[i]; f[8 + i] = p1[i]; }
  return f;
}

// Async copy 16B global -> LDS (CDNA5 ASYNCcnt path)
__device__ __forceinline__ void async_ld_b128(uint32_t lds_byte_off, const __bf16* gptr) {
  asm volatile("global_load_async_to_lds_b128 %0, %1, off"
               :: "v"(lds_byte_off), "v"(gptr) : "memory");
}
__device__ __forceinline__ void wait_async0() {
#if __has_builtin(__builtin_amdgcn_s_wait_asynccnt)
  __builtin_amdgcn_s_wait_asynccnt(0);
#else
  asm volatile("s_wait_asynccnt 0x0" ::: "memory");
#endif
}
__device__ __forceinline__ uint32_t lds_off(const void* p) {
  return (uint32_t)(uintptr_t)p;   // low 32 bits of generic addr = LDS offset
}

// ============================================================================
// K1: q/k/v projections.  D[d, pos] = sum_c W[d,c] * x[n,c,pos] + b[d]
// A = W tile (16d x 32c), B = x tile (32c x 16pos).  Output stored pos-major bf16.
// ============================================================================
__global__ __launch_bounds__(256) void k_proj(
    const float* __restrict__ x,
    const float* __restrict__ Wq, const float* __restrict__ bq,
    const float* __restrict__ Wk, const float* __restrict__ bk,
    const float* __restrict__ Wv, const float* __restrict__ bv,
    __bf16* __restrict__ q_ws, __bf16* __restrict__ k_ws, __bf16* __restrict__ v_ws)
{
  const int lane = threadIdx.x & 31;
  const int wave = threadIdx.x >> 5;
  const int half = lane >> 4;
  const int nn   = lane & 15;

  const int ptile = blockIdx.x;
  const int n     = ptile / (THW / 16);
  const int pos0  = (ptile % (THW / 16)) * 16;
  const int mt    = blockIdx.y * 8 + wave;      // 0..23

  const float* Wmat; const float* bias; __bf16* out; int Dd; int d0;
  if (mt < 4)       { Wmat = Wq; bias = bq; out = q_ws; Dd = CI; d0 = mt * 16; }
  else if (mt < 8)  { Wmat = Wk; bias = bk; out = k_ws; Dd = CI; d0 = (mt - 4) * 16; }
  else              { Wmat = Wv; bias = bv; out = v_ws; Dd = CC; d0 = (mt - 8) * 16; }

  const float* __restrict__ xcol = x + (size_t)n * CC * THW + pos0 + nn;
  const float* __restrict__ wrow = Wmat + (size_t)(d0 + nn) * CC;

  v8f acc = {};
  for (int c0 = 0; c0 < CC; c0 += 32) {
    if (c0 + 32 < CC)
      __builtin_prefetch(xcol + (size_t)(c0 + 32) * THW, 0, 1);  // global_prefetch_b8
    v16bf a, b;
#pragma unroll
    for (int i = 0; i < 16; ++i) {
      const int k = c0 + kidx(i, half);
      a[i] = (__bf16)wrow[k];
      b[i] = (__bf16)xcol[(size_t)k * THW];
    }
    acc = wmma_bf16(a, b, acc);
  }

  __bf16 o[8];
#pragma unroll
  for (int j = 0; j < 8; ++j)
    o[j] = (__bf16)(acc[j] + bias[d0 + 8 * half + j]);
  __bf16* dst = out + ((size_t)n * THW + pos0 + nn) * Dd + d0 + 8 * half;
  *(uint4*)dst = *(const uint4*)o;
}

// ============================================================================
// K2a: E_t[s,t] at each (n,h,w).  16x16, K=64.  8 waves/block.
// ============================================================================
__global__ __launch_bounds__(256) void k_energy_t(
    const __bf16* __restrict__ q_ws, const __bf16* __restrict__ k_ws,
    float* __restrict__ e_ws)
{
  const int lane = threadIdx.x & 31, wave = threadIdx.x >> 5;
  const int half = lane >> 4, nn = lane & 15;
  const int idx = blockIdx.x * 8 + wave;
  const int n = idx / HWD, hw = idx % HWD;

  const __bf16* kb = k_ws + ((size_t)n * THW + hw) * CI;
  const __bf16* qb = q_ws + ((size_t)n * THW + hw) * CI;

  v8f acc = {};
#pragma unroll
  for (int c0 = 0; c0 < CI; c0 += 32) {
    v16bf a = load_row_frag(kb + (size_t)nn * HWD * CI + c0, half);
    v16bf b = load_row_frag(qb + (size_t)nn * HWD * CI + c0, half);
    acc = wmma_bf16(a, b, acc);
  }
#pragma unroll
  for (int j = 0; j < 8; ++j) {
    const int s = j + 8 * half, t = nn;
    e_ws[((size_t)(n * SS + s) * TT + t) * HWD + hw] = acc[j];
  }
}

// ============================================================================
// K2b: E_h (96x96, K=64) at each (n,t,w), diag -> NEG.
// One WG per (n,t,w): async-stage k/q rows (12KB each) into LDS, 6 waves x 6 tiles.
// ============================================================================
__global__ __launch_bounds__(192) void k_energy_h(
    const __bf16* __restrict__ q_ws, const __bf16* __restrict__ k_ws,
    float* __restrict__ e_ws)
{
  __shared__ __bf16 kbuf[HH * CI];   // 12 KB
  __shared__ __bf16 qbuf[HH * CI];   // 12 KB
  const int tid = threadIdx.x;
  const int lane = tid & 31, wave = tid >> 5;
  const int half = lane >> 4, nn = lane & 15;
  const int idx = blockIdx.x;
  const int n = idx / (TT * WW); int r = idx % (TT * WW);
  const int t = r / WW, w = r % WW;

  const size_t base = (size_t)(n * TT + t) * HH;
  const __bf16* kg = k_ws + (base * WW + w) * CI;   // + row*WW*CI + c
  const __bf16* qg = q_ws + (base * WW + w) * CI;

  // 768 16B chunks per buffer, 192 threads x 4 each
#pragma unroll
  for (int i = 0; i < 4; ++i) {
    const int chunk = tid + i * 192;           // 0..767
    const int row = chunk >> 3, c16 = (chunk & 7) * 8;
    const size_t goff = (size_t)row * WW * CI + c16;
    async_ld_b128(lds_off(&kbuf[row * CI + c16]), kg + goff);
    async_ld_b128(lds_off(&qbuf[row * CI + c16]), qg + goff);
  }
  wait_async0();
  __syncthreads();

  const int h0 = wave * 16;
  const v16bf b0 = load_row_frag(&qbuf[(h0 + nn) * CI + 0],  half);
  const v16bf b1 = load_row_frag(&qbuf[(h0 + nn) * CI + 32], half);
#pragma unroll
  for (int st = 0; st < 6; ++st) {
    const int s0 = st * 16;
    v16bf a0 = load_row_frag(&kbuf[(s0 + nn) * CI + 0],  half);
    v16bf a1 = load_row_frag(&kbuf[(s0 + nn) * CI + 32], half);
    v8f acc = {};
    acc = wmma_bf16(a0, b0, acc);
    acc = wmma_bf16(a1, b1, acc);
#pragma unroll
    for (int j = 0; j < 8; ++j) {
      const int s = s0 + j + 8 * half, h = h0 + nn;
      const float v = (s == h) ? NEGV : acc[j];
      e_ws[((size_t)(n * SS + TT + s) * TT + t) * HWD + h * WW + w] = v;
    }
  }
}

// ============================================================================
// K2c: E_w (96x96, K=64) at each (n,t,h), diag -> NEG.  q/k block is fully
// contiguous (12KB) -> linear async stage.
// ============================================================================
__global__ __launch_bounds__(192) void k_energy_w(
    const __bf16* __restrict__ q_ws, const __bf16* __restrict__ k_ws,
    float* __restrict__ e_ws)
{
  __shared__ __bf16 kbuf[WW * CI];
  __shared__ __bf16 qbuf[WW * CI];
  const int tid = threadIdx.x;
  const int lane = tid & 31, wave = tid >> 5;
  const int half = lane >> 4, nn = lane & 15;
  const int idx = blockIdx.x;
  const int n = idx / (TT * HH); int r = idx % (TT * HH);
  const int t = r / HH, h = r % HH;

  const size_t rowb = ((size_t)(n * TT + t) * HH + h) * WW;
  const __bf16* kg = k_ws + rowb * CI;   // 6144 contiguous bf16
  const __bf16* qg = q_ws + rowb * CI;

#pragma unroll
  for (int i = 0; i < 4; ++i) {
    const int chunk = tid + i * 192;           // 0..767 chunks of 8 bf16
    async_ld_b128(lds_off(&kbuf[chunk * 8]), kg + chunk * 8);
    async_ld_b128(lds_off(&qbuf[chunk * 8]), qg + chunk * 8);
  }
  wait_async0();
  __syncthreads();

  const int w0 = wave * 16;
  const v16bf b0 = load_row_frag(&qbuf[(w0 + nn) * CI + 0],  half);
  const v16bf b1 = load_row_frag(&qbuf[(w0 + nn) * CI + 32], half);
#pragma unroll
  for (int st = 0; st < 6; ++st) {
    const int s0 = st * 16;
    v16bf a0 = load_row_frag(&kbuf[(s0 + nn) * CI + 0],  half);
    v16bf a1 = load_row_frag(&kbuf[(s0 + nn) * CI + 32], half);
    v8f acc = {};
    acc = wmma_bf16(a0, b0, acc);
    acc = wmma_bf16(a1, b1, acc);
#pragma unroll
    for (int j = 0; j < 8; ++j) {
      const int s = s0 + j + 8 * half, w = w0 + nn;
      const float v = (s == w) ? NEGV : acc[j];
      e_ws[((size_t)(n * SS + TT + HH + s) * TT + t) * HWD + h * WW + w] = v;
    }
  }
}

// ============================================================================
// K3: in-place softmax over the 208 criss-cross scores of each query.
// ============================================================================
__global__ __launch_bounds__(256) void k_softmax(float* __restrict__ e)
{
  const size_t q = (size_t)blockIdx.x * blockDim.x + threadIdx.x;
  const int n = (int)(q / THW);
  const size_t thw = q % THW;
  float* p = e + (size_t)n * SS * THW + thw;
  const size_t st = THW;
  float mx = -3.4e38f;
  for (int s = 0; s < SS; ++s) mx = fmaxf(mx, p[s * st]);
  float sum = 0.f;
  for (int s = 0; s < SS; ++s) { float ex = __expf(p[s * st] - mx); p[s * st] = ex; sum += ex; }
  const float inv = 1.f / sum;
  for (int s = 0; s < SS; ++s) p[s * st] *= inv;
}

// ============================================================================
// K4a: out[c,t] = sum_s a_t[s,t] v[c,s] at each (n,h,w). K=16 (padded to 32).
// 16 waves/block = 16 c-tiles.  Plain store initializes d_out.
// ============================================================================
__global__ __launch_bounds__(512) void k_agg_t(
    const __bf16* __restrict__ v_ws, const float* __restrict__ a_ws,
    float* __restrict__ out)
{
  const int lane = threadIdx.x & 31, wave = threadIdx.x >> 5;
  const int half = lane >> 4, nn = lane & 15;
  const int idx = blockIdx.x;
  const int n = idx / HWD, hw = idx % HWD;
  const int c0 = wave * 16;

  v16bf a, b;
#pragma unroll
  for (int i = 0; i < 16; ++i) {
    const int s = kidx(i, half);
    if (s < TT) {
      a[i] = v_ws[((size_t)(n * TT + s) * HWD + hw) * CC + c0 + nn];
      b[i] = (__bf16)a_ws[((size_t)(n * SS + s) * TT + nn) * HWD + hw];
    } else { a[i] = (__bf16)0.f; b[i] = (__bf16)0.f; }
  }
  v8f acc = {};
  acc = wmma_bf16(a, b, acc);
#pragma unroll
  for (int j = 0; j < 8; ++j) {
    const int c = c0 + j + 8 * half, t = nn;
    out[((size_t)(n * CC + c) * TT + t) * HWD + hw] = acc[j];
  }
}

// ============================================================================
// K4b: out[c,h] += sum_s a_h[s,h] v[n,c,t,s,w] at each (n,t,w).  K=96.
// One 512-thread WG per (n,t,w): stage V transposed [c][s] (48KB) and the
// attention slice [h][s] (18KB, bf16) in LDS; 16 waves = c-tiles, 6 h-tiles ea.
// ============================================================================
__global__ __launch_bounds__(512) void k_agg_h(
    const __bf16* __restrict__ v_ws, const float* __restrict__ a_ws,
    float* __restrict__ out)
{
  __shared__ __bf16 vbuf[CC * HH];   // [c][s]  49152 B
  __shared__ __bf16 abuf[HH * HH];   // [h][s]  18432 B
  const int tid = threadIdx.x;
  const int lane = tid & 31, wave = tid >> 5;
  const int half = lane >> 4, nn = lane & 15;
  const int idx = blockIdx.x;
  const int n = idx / (TT * WW); int r = idx % (TT * WW);
  const int t = r / WW, w = r % WW;

  const __bf16* vg = v_ws + (((size_t)(n * TT + t) * HH) * WW + w) * CC; // + s*WW*CC + c
  const float*  ag = a_ws + ((size_t)(n * SS + TT) * TT + t) * HWD + w;  // + s*TT*HWD + h*WW

  // stage V transposed: 3072 chunks of (s, 8 c), 6 per thread
#pragma unroll
  for (int i = 0; i < 6; ++i) {
    const int q = tid + i * 512;               // 0..3071
    const int s = q >> 5, c8 = (q & 31) * 8;
    v8bf pk = *(const v8bf*)(vg + (size_t)s * WW * CC + c8);
#pragma unroll
    for (int j = 0; j < 8; ++j) vbuf[(c8 + j) * HH + s] = pk[j];
  }
  // stage attention (f32 -> bf16): 9216 elems, 18 per thread
#pragma unroll
  for (int i = 0; i < 18; ++i) {
    const int e = tid + i * 512;               // 0..9215
    const int h = e % HH, s = e / HH;
    abuf[h * HH + s] = (__bf16)ag[(size_t)s * TT * HWD + (size_t)h * WW];
  }
  __syncthreads();

  const int c0 = wave * 16;
  const v16bf a0 = load_row_frag(&vbuf[(c0 + nn) * HH + 0],  half);
  const v16bf a1 = load_row_frag(&vbuf[(c0 + nn) * HH + 32], half);
  const v16bf a2 = load_row_frag(&vbuf[(c0 + nn) * HH + 64], half);
#pragma unroll
  for (int ht = 0; ht < 6; ++ht) {
    const int h0 = ht * 16;
    v16bf b0 = load_row_frag(&abuf[(h0 + nn) * HH + 0],  half);
    v16bf b1 = load_row_frag(&abuf[(h0 + nn) * HH + 32], half);
    v16bf b2 = load_row_frag(&abuf[(h0 + nn) * HH + 64], half);
    v8f acc = {};
    acc = wmma_bf16(a0, b0, acc);
    acc = wmma_bf16(a1, b1, acc);
    acc = wmma_bf16(a2, b2, acc);
#pragma unroll
    for (int j = 0; j < 8; ++j) {
      const int c = c0 + j + 8 * half, h = h0 + nn;
      const size_t o = ((size_t)(n * CC + c) * TT + t) * HWD + h * WW + w;
      out[o] = out[o] + acc[j];
    }
  }
}

// ============================================================================
// K4c: out[c,w] += sum_s a_w[s,w] v[n,c,t,h,s]; final *= gamma.  K=96.
// Same LDS structure as K4b; V rows here are fully contiguous.
// ============================================================================
__global__ __launch_bounds__(512) void k_agg_w(
    const __bf16* __restrict__ v_ws, const float* __restrict__ a_ws,
    float* __restrict__ out, const float* __restrict__ gamma)
{
  __shared__ __bf16 vbuf[CC * WW];   // [c][s]
  __shared__ __bf16 abuf[WW * WW];   // [w][s]
  const int tid = threadIdx.x;
  const int lane = tid & 31, wave = tid >> 5;
  const int half = lane >> 4, nn = lane & 15;
  const int idx = blockIdx.x;
  const int n = idx / (TT * HH); int r = idx % (TT * HH);
  const int t = r / HH, h = r % HH;
  const float g = gamma[0];

  const size_t rowb = ((size_t)(n * TT + t) * HH + h) * WW;
  const __bf16* vg = v_ws + rowb * CC;                                    // + s*CC + c
  const float*  ag = a_ws + ((size_t)(n * SS + TT + HH) * TT + t) * HWD + h * WW; // + s*TT*HWD + w

#pragma unroll
  for (int i = 0; i < 6; ++i) {
    const int q = tid + i * 512;
    const int s = q >> 5, c8 = (q & 31) * 8;
    v8bf pk = *(const v8bf*)(vg + (size_t)s * CC + c8);
#pragma unroll
    for (int j = 0; j < 8; ++j) vbuf[(c8 + j) * WW + s] = pk[j];
  }
#pragma unroll
  for (int i = 0; i < 18; ++i) {
    const int e = tid + i * 512;
    const int w = e % WW, s = e / WW;     // consecutive tid -> contiguous global reads
    abuf[w * WW + s] = (__bf16)ag[(size_t)s * TT * HWD + w];
  }
  __syncthreads();

  const int c0 = wave * 16;
  const v16bf a0 = load_row_frag(&vbuf[(c0 + nn) * WW + 0],  half);
  const v16bf a1 = load_row_frag(&vbuf[(c0 + nn) * WW + 32], half);
  const v16bf a2 = load_row_frag(&vbuf[(c0 + nn) * WW + 64], half);
#pragma unroll
  for (int wt = 0; wt < 6; ++wt) {
    const int w0 = wt * 16;
    v16bf b0 = load_row_frag(&abuf[(w0 + nn) * WW + 0],  half);
    v16bf b1 = load_row_frag(&abuf[(w0 + nn) * WW + 32], half);
    v16bf b2 = load_row_frag(&abuf[(w0 + nn) * WW + 64], half);
    v8f acc = {};
    acc = wmma_bf16(a0, b0, acc);
    acc = wmma_bf16(a1, b1, acc);
    acc = wmma_bf16(a2, b2, acc);
#pragma unroll
    for (int j = 0; j < 8; ++j) {
      const int c = c0 + j + 8 * half, w = w0 + nn;
      const size_t o = ((size_t)(n * CC + c) * TT + t) * HWD + h * WW + w;
      out[o] = g * (out[o] + acc[j]);
    }
  }
}

// ============================================================================
extern "C" void kernel_launch(void* const* d_in, const int* in_sizes, int n_in,
                              void* d_out, int out_size, void* d_ws, size_t ws_size,
                              hipStream_t stream)
{
  const float* x     = (const float*)d_in[0];
  const float* Wq    = (const float*)d_in[1];
  const float* bq    = (const float*)d_in[2];
  const float* Wk    = (const float*)d_in[3];
  const float* bk    = (const float*)d_in[4];
  const float* Wv    = (const float*)d_in[5];
  const float* bv    = (const float*)d_in[6];
  const float* gamma = (const float*)d_in[7];
  float* out = (float*)d_out;

  // ws layout: q bf16 (37.7MB) | k bf16 (37.7MB) | v bf16 (151MB) | energy f32 (245MB)
  char* ws = (char*)d_ws;
  __bf16* q_ws = (__bf16*)(ws);
  __bf16* k_ws = (__bf16*)(ws + (size_t)37748736);
  __bf16* v_ws = (__bf16*)(ws + (size_t)75497472);
  float*  e_ws = (float*) (ws + (size_t)226492416);

  k_proj<<<dim3(NB * THW / 16, 3), 256, 0, stream>>>(x, Wq, bq, Wk, bk, Wv, bv,
                                                     q_ws, k_ws, v_ws);
  k_energy_t<<<dim3(NB * HWD / 8), 256, 0, stream>>>(q_ws, k_ws, e_ws);
  k_energy_h<<<dim3(NB * TT * WW), 192, 0, stream>>>(q_ws, k_ws, e_ws);
  k_energy_w<<<dim3(NB * TT * HH), 192, 0, stream>>>(q_ws, k_ws, e_ws);
  k_softmax<<<dim3(NB * THW / 256), 256, 0, stream>>>(e_ws);
  k_agg_t<<<dim3(NB * HWD), 512, 0, stream>>>(v_ws, e_ws, out);
  k_agg_h<<<dim3(NB * TT * WW), 512, 0, stream>>>(v_ws, e_ws, out);
  k_agg_w<<<dim3(NB * TT * HH), 512, 0, stream>>>(v_ws, e_ws, out, gamma);
}
